// RecurrentGemma_10015863734593
// MI455X (gfx1250) — matrix operand order repaired and checked by reading, not compiled, not run on hardware
//
#include <hip/hip_runtime.h>
#include <hip/hip_bf16.h>

// ---------------------------------------------------------------------------
// RecurrentGemma forward -> scalar NLL, for MI455X (gfx1250, wave32, WMMA).
// All dense matmuls use a pipelined WMMA f16-in/f32-acc tiled GEMM with
// guard-free vectorized loads (M%32==0, N%64==0, K%32==0 guaranteed by host).
// ---------------------------------------------------------------------------

#define B_   8
#define T_   1024
#define V_   256
#define W_   256
#define LRU_ 256
#define H_   8
#define HD_  32
#define EXP_ 768
#define BT_  (B_ * T_)
#define WIN_ 256

typedef __attribute__((ext_vector_type(16))) _Float16 v16h;
typedef __attribute__((ext_vector_type(8)))  _Float16 v8h;
typedef __attribute__((ext_vector_type(4)))  _Float16 v4h;
typedef __attribute__((ext_vector_type(8)))  float    v8f;

union FragAB { v16h v; v8h h8[2]; _Float16 h[16]; };
union FragC  { v8f  v; float f[8]; };

// ------------------------------- device helpers ----------------------------

__device__ __forceinline__ float gelu_tanh(float x) {
  const float c0 = 0.7978845608028654f;  // sqrt(2/pi)
  float x3 = x * x * x;
  return 0.5f * x * (1.0f + tanhf(c0 * (x + 0.044715f * x3)));
}
__device__ __forceinline__ float sigmoidf(float x) { return 1.0f / (1.0f + expf(-x)); }
__device__ __forceinline__ float softplusf(float x) { return (x > 20.0f) ? x : log1pf(expf(x)); }

__device__ __forceinline__ float waveSum(float v) {
  #pragma unroll
  for (int off = 16; off > 0; off >>= 1) v += __shfl_xor(v, off, 32);
  return v;
}
__device__ __forceinline__ float waveMax(float v) {
  #pragma unroll
  for (int off = 16; off > 0; off >>= 1) v = fmaxf(v, __shfl_xor(v, off, 32));
  return v;
}

// ------------------------------- WMMA GEMM ---------------------------------
// C[M,N] = act(A[M,K] @ B[K,N] + bias[N]).  transB: B stored [N,K].
// Requirements: M%32==0, N%64==0, K%32==0 (host guarantees).
// Block: 256 threads = 8 waves; block tile 32(M) x 64(N); K-step 32.
// LDS: As row-major (k-contiguous), Bs n-major (k-contiguous) so both
// fragments are two contiguous 16B ds loads per lane.
__global__ void gemm_wmma_kernel(const float* __restrict__ A,
                                 const float* __restrict__ Bm,
                                 const float* __restrict__ bias,
                                 float* __restrict__ C,
                                 int M, int N, int K, int act, int transB) {
  __shared__ _Float16 As[32][40];   // [m][k], padded row stride (80B)
  __shared__ _Float16 Bs[64][40];   // [n][k], padded row stride (80B)

  const int tid  = threadIdx.x;
  const int lane = tid & 31;
  const int wave = tid >> 5;
  const int hl   = lane >> 4;       // lane half (0/1)
  const int l16  = lane & 15;
  const int wm   = (wave >> 2) * 16;
  const int wn   = (wave & 3) * 16;

  const int rowBase = blockIdx.y * 32;
  const int colBase = blockIdx.x * 64;

  // Per-thread tile-fill coordinates.
  const int ar = tid >> 3;          // A row 0..31
  const int ac = (tid & 7) * 4;     // A k-col 0..28 step 4
  // B fill: 512 float4 = 256 threads x 2 iterations.

  float4 aReg;
  float4 bReg[2];

  auto loadTile = [&](int k0) {
    aReg = *(const float4*)&A[(rowBase + ar) * K + k0 + ac];
    if (!transB) {
      #pragma unroll
      for (int it = 0; it < 2; ++it) {
        int f  = tid + 256 * it;
        int kk = f >> 4, n4 = (f & 15) * 4;
        bReg[it] = *(const float4*)&Bm[(k0 + kk) * N + colBase + n4];
      }
    } else {
      #pragma unroll
      for (int it = 0; it < 2; ++it) {
        int f = tid + 256 * it;
        int n = f >> 3, k4 = (f & 7) * 4;
        bReg[it] = *(const float4*)&Bm[(colBase + n) * K + k0 + k4];
      }
    }
  };

  auto stageTile = [&]() {
    v4h a4 = { (_Float16)aReg.x, (_Float16)aReg.y, (_Float16)aReg.z, (_Float16)aReg.w };
    *(v4h*)&As[ar][ac] = a4;
    if (!transB) {
      #pragma unroll
      for (int it = 0; it < 2; ++it) {
        int f  = tid + 256 * it;
        int kk = f >> 4, n4 = (f & 15) * 4;
        Bs[n4 + 0][kk] = (_Float16)bReg[it].x;
        Bs[n4 + 1][kk] = (_Float16)bReg[it].y;
        Bs[n4 + 2][kk] = (_Float16)bReg[it].z;
        Bs[n4 + 3][kk] = (_Float16)bReg[it].w;
      }
    } else {
      #pragma unroll
      for (int it = 0; it < 2; ++it) {
        int f = tid + 256 * it;
        int n = f >> 3, k4 = (f & 7) * 4;
        v4h b4 = { (_Float16)bReg[it].x, (_Float16)bReg[it].y,
                   (_Float16)bReg[it].z, (_Float16)bReg[it].w };
        *(v4h*)&Bs[n][k4] = b4;
      }
    }
  };

  FragC acc;
  #pragma unroll
  for (int r = 0; r < 8; ++r) acc.f[r] = 0.0f;

  const int nk = K >> 5;
  loadTile(0);
  for (int s = 0; s < nk; ++s) {
    stageTile();
    __syncthreads();
    if (s + 1 < nk) loadTile((s + 1) << 5);  // overlap next-tile loads with math

    // Fragments per ISA 7.12.2 (16-bit A 16x32; B 32x16; wave32):
    //  A: elems 0-7 -> K = hl*8 + e ; elems 8-15 -> K = 16 + hl*8 + (e-8)
    //  B: elems 0-7 -> K = hl*8 + e ; elems 8-15 -> K = 16 + hl*8 + (e-8) at fixed n
    FragAB fa, fb;
    fa.h8[0] = *(const v8h*)&As[wm + l16][hl * 8];
    fa.h8[1] = *(const v8h*)&As[wm + l16][16 + hl * 8];
    fb.h8[0] = *(const v8h*)&Bs[wn + l16][hl * 8];
    fb.h8[1] = *(const v8h*)&Bs[wn + l16][16 + hl * 8];
    acc.v = __builtin_amdgcn_wmma_f32_16x16x32_f16(
        false, fa.v, false, fb.v, (short)0, acc.v, false, false);
    __syncthreads();
  }

  // Store C: lane half selects M rows r+8*hl, lane&15 selects N col.
  const int n  = colBase + wn + l16;
  const float bv = bias ? bias[n] : 0.0f;
  #pragma unroll
  for (int r = 0; r < 8; ++r) {
    int m = rowBase + wm + r + 8 * hl;
    float v = acc.f[r] + bv;
    if (act == 1) v = gelu_tanh(v);
    C[m * N + n] = v;
  }
}

// ------------------------------- pointwise kernels -------------------------

__global__ void embed_kernel(const int* __restrict__ x_raw,
                             const float* __restrict__ embed,
                             float* __restrict__ x) {
  int bt = blockIdx.x, c = threadIdx.x;
  int b = bt >> 10, t = bt & 1023;
  int id = (t == 0) ? 128 : x_raw[b * T_ + (t - 1)];
  x[bt * W_ + c] = embed[id * W_ + c] * 16.0f;  // sqrt(256)
}

__global__ void rmsnorm_kernel(const float* __restrict__ x,
                               const float* __restrict__ scale,
                               float* __restrict__ out) {
  __shared__ float red[8];
  int bt = blockIdx.x, c = threadIdx.x;
  int wave = c >> 5, lane = c & 31;
  float v = x[bt * W_ + c];
  float s = waveSum(v * v);
  if (lane == 0) red[wave] = s;
  __syncthreads();
  float tot = red[0] + red[1] + red[2] + red[3] + red[4] + red[5] + red[6] + red[7];
  float r = rsqrtf(tot * (1.0f / W_) + 1e-6f);
  out[bt * W_ + c] = v * r * (1.0f + scale[c]);
}

__global__ void add_kernel(float* __restrict__ x, const float* __restrict__ y, int n) {
  int i = blockIdx.x * blockDim.x + threadIdx.x;
  if (i < n) x[i] += y[i];
}

__global__ void gelumul_kernel(float* __restrict__ a, const float* __restrict__ b, int n) {
  int i = blockIdx.x * blockDim.x + threadIdx.x;
  if (i < n) a[i] = gelu_tanh(a[i]) * b[i];
}

__global__ void pack_kv_kernel(const float* __restrict__ kw,
                               const float* __restrict__ vw,
                               float* __restrict__ wkv) {
  int i = blockIdx.x * blockDim.x + threadIdx.x;  // W*64
  int w = i >> 6, j = i & 63;
  wkv[i] = (j < HD_) ? kw[w * HD_ + j] : vw[w * HD_ + (j - HD_)];
}

// ------------------------------- recurrent block ---------------------------

__global__ void conv_kernel(const float* __restrict__ u_pre,
                            const float* __restrict__ cw,
                            const float* __restrict__ cb,
                            float* __restrict__ u) {
  int idx = blockIdx.x * blockDim.x + threadIdx.x;  // BT * LRU
  int c = idx & 255, bt = idx >> 8, t = bt & 1023;
  float acc = cb[c];
  #pragma unroll
  for (int i = 0; i < 4; ++i) {
    int tt = t - 3 + i;
    if (tt >= 0) acc += u_pre[(bt - (3 - i)) * LRU_ + c] * cw[i * LRU_ + c];
  }
  u[idx] = acc;
}

__global__ void gates_kernel(const float* __restrict__ u,
                             const float* __restrict__ igw, const float* __restrict__ igb,
                             const float* __restrict__ agw, const float* __restrict__ agb,
                             const float* __restrict__ ap,
                             float* __restrict__ a_out, float* __restrict__ xn_out) {
  int idx = blockIdx.x * blockDim.x + threadIdx.x;  // BT * LRU
  int c = idx & 255, bt = idx >> 8, t = bt & 1023;
  int h = c >> 5, e = c & 31;
  const float* uh = u + bt * LRU_ + h * HD_;
  float gx = igb[h * HD_ + e], ga = agb[h * HD_ + e];
  #pragma unroll 8
  for (int d = 0; d < HD_; ++d) {
    float uv = uh[d];
    gx += uv * igw[(h * HD_ + d) * HD_ + e];
    ga += uv * agw[(h * HD_ + d) * HD_ + e];
  }
  gx = sigmoidf(gx);
  ga = sigmoidf(ga);
  float log_a = -8.0f * ga * softplusf(ap[c]);
  float a, mult;
  if (t == 0) { a = 0.0f; mult = 1.0f; }
  else {
    a = expf(log_a);
    mult = sqrtf(fmaxf(0.0f, 1.0f - expf(2.0f * log_a)));
  }
  xn_out[idx] = u[idx] * gx * mult;
  a_out[idx]  = a;
}

__global__ void scan_kernel(const float* __restrict__ a, const float* __restrict__ xn,
                            const float* __restrict__ y, float* __restrict__ prod) {
  int idx = blockIdx.x * blockDim.x + threadIdx.x;  // B * LRU = 2048
  int b = idx >> 8, c = idx & 255;
  float h = 0.0f;
  for (int t = 0; t < T_; ++t) {
    int o = (b * T_ + t) * LRU_ + c;
    h = a[o] * h + xn[o];
    prod[o] = h * y[o];
  }
}

// ------------------------------- attention ---------------------------------

__global__ void rope_q_kernel(float* __restrict__ q) {  // [BT, H, HD]; also scales q
  int bt = blockIdx.x, tid = threadIdx.x;
  int h = tid >> 5, d = tid & 31;
  int t = bt & 1023;
  float* qp = q + bt * (H_ * HD_) + h * HD_;
  int j = d & 15;
  float ts = powf(10000.0f, (2.0f * j) / (float)HD_);
  float ang = (float)t / ts;
  float sn = sinf(ang), cs = cosf(ang);
  float av = qp[j], bv = qp[j + 16];
  float val = (d < 16) ? (av * cs - bv * sn) : (bv * cs + av * sn);
  __syncthreads();
  qp[d] = val * 0.17677669529663687f;  // HD^-0.5
}

__global__ void rope_k_kernel(float* __restrict__ kv) {  // [BT, 64]; k part only
  int bt = blockIdx.x, d = threadIdx.x;                   // 32 threads
  int t = bt & 1023;
  float* kp = kv + bt * 64;
  int j = d & 15;
  float ts = powf(10000.0f, (2.0f * j) / (float)HD_);
  float ang = (float)t / ts;
  float sn = sinf(ang), cs = cosf(ang);
  float av = kp[j], bv = kp[j + 16];
  float val = (d < 16) ? (av * cs - bv * sn) : (bv * cs + av * sn);
  __syncthreads();
  kp[d] = val;
}

// One block per token; wave = head; lanes parallelize HD; online softmax.
// kv: [BT, 64], cols 0-31 = roped k, 32-63 = v.
__global__ void attn_kernel(const float* __restrict__ q, const float* __restrict__ kv,
                            float* __restrict__ out) {
  int bt = blockIdx.x, b = bt >> 10, t = bt & 1023;
  int h = threadIdx.x >> 5, d = threadIdx.x & 31;
  float qd = q[bt * (H_ * HD_) + h * HD_ + d];
  float m = -1e30f, l = 0.0f, acc = 0.0f;
  int s0 = t - (WIN_ - 1); if (s0 < 0) s0 = 0;
  for (int s = s0; s <= t; ++s) {
    const float* kvp = kv + (b * T_ + s) * 64;
    float logit = waveSum(qd * kvp[d]);
    float mn = fmaxf(m, logit);
    float sc = expf(m - mn);
    float p  = expf(logit - mn);
    l   = l * sc + p;
    acc = acc * sc + p * kvp[32 + d];
    m = mn;
  }
  out[bt * (H_ * HD_) + h * HD_ + d] = acc / l;
}

// ------------------------------- loss --------------------------------------

__global__ void init_acc_kernel(float* acc) { acc[0] = 0.0f; }

__global__ void loss_kernel(const float* __restrict__ logits,
                            const int* __restrict__ x_raw,
                            const int* __restrict__ lengths,
                            float* __restrict__ acc) {
  __shared__ float red[8];
  int bt = blockIdx.x, b = bt >> 10, t = bt & 1023, c = threadIdx.x;
  int wave = c >> 5, lane = c & 31;
  float z = 30.0f * tanhf(logits[bt * V_ + c] * (1.0f / 30.0f));
  float wm = waveMax(z);
  if (lane == 0) red[wave] = wm;
  __syncthreads();
  float mx = fmaxf(fmaxf(fmaxf(red[0], red[1]), fmaxf(red[2], red[3])),
                   fmaxf(fmaxf(red[4], red[5]), fmaxf(red[6], red[7])));
  __syncthreads();
  float ex = expf(z - mx);
  float ws = waveSum(ex);
  if (lane == 0) red[wave] = ws;
  __syncthreads();
  float tot = red[0] + red[1] + red[2] + red[3] + red[4] + red[5] + red[6] + red[7];
  int tgt = x_raw[bt];
  if (c == tgt && t < lengths[b]) {
    atomicAdd(acc, (z - mx) - logf(tot));
  }
}

__global__ void finalize_kernel(const float* __restrict__ acc,
                                const int* __restrict__ lengths,
                                float* __restrict__ out) {
  float nl = 0.0f;
  for (int b = 0; b < B_; ++b) nl += (float)lengths[b];
  out[0] = -acc[0] / (nl * 0.6931471805599453f);
}

// ------------------------------- host orchestration ------------------------

extern "C" void kernel_launch(void* const* d_in, const int* in_sizes, int n_in,
                              void* d_out, int out_size, void* d_ws, size_t ws_size,
                              hipStream_t stream) {
  (void)in_sizes; (void)n_in; (void)out_size; (void)ws_size;

  const float* embed      = (const float*)d_in[0];
  const float* tnorm      = (const float*)d_in[1];
  const float* cnorm      = (const float*)d_in[2];
  const float* rec_wy     = (const float*)d_in[3];
  const float* rec_by     = (const float*)d_in[4];
  const float* rec_wx     = (const float*)d_in[5];
  const float* rec_bx     = (const float*)d_in[6];
  const float* conv_w     = (const float*)d_in[7];
  const float* conv_b     = (const float*)d_in[8];
  const float* a_param    = (const float*)d_in[9];
  const float* igate_w    = (const float*)d_in[10];
  const float* igate_b    = (const float*)d_in[11];
  const float* agate_w    = (const float*)d_in[12];
  const float* agate_b    = (const float*)d_in[13];
  const float* rec_wo     = (const float*)d_in[14];
  const float* rec_bo     = (const float*)d_in[15];
  const float* attn_qw    = (const float*)d_in[16];
  const float* attn_kw    = (const float*)d_in[17];
  const float* attn_vw    = (const float*)d_in[18];
  const float* attn_ow    = (const float*)d_in[19];
  const float* attn_ob    = (const float*)d_in[20];
  const float* mlp_up_w   = (const float*)d_in[21];
  const float* mlp_up_b   = (const float*)d_in[22];
  const float* mlp_down_w = (const float*)d_in[23];
  const float* mlp_down_b = (const float*)d_in[24];
  const float* fnorm      = (const float*)d_in[25];
  const int*   x_raw      = (const int*)d_in[26];
  const int*   lengths    = (const int*)d_in[27];

  // Workspace layout (floats). Total ~96 MB.
  float* ws   = (float*)d_ws;
  float* x    = ws;                   // BT*W
  float* h    = x  + BT_ * W_;        // BT*W
  float* t2   = h  + BT_ * W_;        // BT*W
  float* t3   = t2 + BT_ * W_;        // BT*W
  float* t4   = t3 + BT_ * W_;        // BT*W
  float* t5   = t4 + BT_ * W_;        // BT*W
  float* t0   = t5 + BT_ * W_;        // BT*EXP
  float* t1   = t0 + BT_ * EXP_;      // BT*EXP
  float* accb = t1 + BT_ * EXP_;      // 1 (+pad)
  float* wkv  = accb + 64;            // W*64 packed k|v weights
  float* kvb  = t0;                   // BT*64 (alias: t0 free during attention)

  auto gemm = [&](const float* A, const float* Bm, const float* bias, float* C,
                  int M, int N, int K, int act, int transB) {
    dim3 g((N + 63) / 64, (M + 31) / 32);
    gemm_wmma_kernel<<<g, 256, 0, stream>>>(A, Bm, bias, C, M, N, K, act, transB);
  };

  embed_kernel<<<BT_, 256, 0, stream>>>(x_raw, embed, x);

  const char types[8] = {'R','R','A','R','R','A','R','R'};
  int ri = 0, ai = 0;
  for (int i = 0; i < 8; ++i) {
    rmsnorm_kernel<<<BT_, 256, 0, stream>>>(x, tnorm + i * W_, h);
    if (types[i] == 'R') {
      gemm(h, rec_wy + ri * W_ * LRU_, rec_by + ri * LRU_, t3, BT_, LRU_, W_, 1, 0); // y=gelu
      gemm(h, rec_wx + ri * W_ * LRU_, rec_bx + ri * LRU_, t2, BT_, LRU_, W_, 0, 0); // u_pre
      conv_kernel<<<BT_ * LRU_ / 256, 256, 0, stream>>>(t2, conv_w + ri * 4 * LRU_,
                                                        conv_b + ri * LRU_, t4);
      gates_kernel<<<BT_ * LRU_ / 256, 256, 0, stream>>>(
          t4, igate_w + ri * H_ * HD_ * HD_, igate_b + ri * H_ * HD_,
          agate_w + ri * H_ * HD_ * HD_, agate_b + ri * H_ * HD_,
          a_param + ri * LRU_, t2, t5);
      scan_kernel<<<B_, 256, 0, stream>>>(t2, t5, t3, t4);
      gemm(t4, rec_wo + ri * LRU_ * W_, rec_bo + ri * W_, t5, BT_, W_, LRU_, 0, 0);
      add_kernel<<<BT_ * W_ / 256, 256, 0, stream>>>(x, t5, BT_ * W_);
      ++ri;
    } else {
      pack_kv_kernel<<<W_ * 64 / 256, 256, 0, stream>>>(
          attn_kw + ai * W_ * HD_, attn_vw + ai * W_ * HD_, wkv);
      gemm(h, attn_qw + ai * W_ * H_ * HD_, nullptr, t2, BT_, H_ * HD_, W_, 0, 0);
      gemm(h, wkv, nullptr, kvb, BT_, 64, W_, 0, 0);  // [BT,64] = k|v
      rope_q_kernel<<<BT_, 256, 0, stream>>>(t2);
      rope_k_kernel<<<BT_, 32, 0, stream>>>(kvb);
      attn_kernel<<<BT_, 256, 0, stream>>>(t2, kvb, t3);
      gemm(t3, attn_ow + ai * H_ * HD_ * W_, attn_ob + ai * W_, t4, BT_, W_, H_ * HD_, 0, 0);
      add_kernel<<<BT_ * W_ / 256, 256, 0, stream>>>(x, t4, BT_ * W_);
      ++ai;
    }
    rmsnorm_kernel<<<BT_, 256, 0, stream>>>(x, cnorm + i * W_, h);
    gemm(h, mlp_up_w + (i * 2 + 0) * W_ * EXP_, mlp_up_b + (i * 2 + 0) * EXP_, t0,
         BT_, EXP_, W_, 0, 0);
    gemm(h, mlp_up_w + (i * 2 + 1) * W_ * EXP_, mlp_up_b + (i * 2 + 1) * EXP_, t1,
         BT_, EXP_, W_, 0, 0);
    gelumul_kernel<<<BT_ * EXP_ / 256, 256, 0, stream>>>(t0, t1, BT_ * EXP_);
    gemm(t0, mlp_down_w + i * EXP_ * W_, mlp_down_b + i * W_, t4, BT_, W_, EXP_, 0, 0);
    add_kernel<<<BT_ * W_ / 256, 256, 0, stream>>>(x, t4, BT_ * W_);
  }

  rmsnorm_kernel<<<BT_, 256, 0, stream>>>(x, fnorm, h);
  gemm(h, embed, nullptr, t2, BT_, V_, W_, 0, 1);  // logits = h @ embed^T

  init_acc_kernel<<<1, 1, 0, stream>>>(accb);
  loss_kernel<<<BT_, 256, 0, stream>>>(t2, x_raw, lengths, accb);
  finalize_kernel<<<1, 1, 0, stream>>>(accb, lengths, (float*)d_out);
}
